// GATrBlock_14843406975354
// MI455X (gfx1250) — matrix operand order repaired and checked by reading, not compiled, not run on hardware
//
#include <hip/hip_runtime.h>
#include <math.h>

// ---------------------------------------------------------------------------
// PGA G(3,0,1) blade metadata (blade order matches the reference _BLADES list)
// ---------------------------------------------------------------------------
__device__ __constant__ int   GRADE[16]   = {0,1,1,1,1,2,2,2,2,2,2,3,3,3,3,4};
__device__ __constant__ int   HASE0[16]   = {0,1,0,0,0,1,1,1,0,0,0,1,1,1,0,1};
__device__ __constant__ int   PARTNER[16] = {0,0,0,0,0,2,3,4,0,0,0,8,9,10,0,14};
__device__ __constant__ float IPF[16]     = {1,0,1,1,1,0,0,0,1,1,1,0,0,0,1,0};

constexpr int BL[16] = {0,1,2,4,8,3,5,9,6,10,12,7,11,13,14,15};
constexpr int idxof(int b){ int r=0; for(int i=0;i<16;i++) if(BL[i]==b) r=i; return r; }
constexpr int pcnt(int x){ int c=0; while(x){ c += x&1; x>>=1; } return c; }
constexpr float sgn(int a,int b){ int s=0; a>>=1; while(a){ s+=pcnt(a&b); a>>=1; } return (s&1)?-1.f:1.f; }

struct SpT { int n; unsigned short e[224]; float s[224]; };

constexpr SpT make_gp(){
  SpT t{}; t.n = 0;
  for(int j=0;j<16;j++) for(int k=0;k<16;k++){
    int a = BL[j], b = BL[k];
    if (a & b & 1) continue;                 // e0 degenerate: e0*e0 = 0
    t.e[t.n] = (unsigned short)(idxof(a^b) | (j<<4) | (k<<8));
    t.s[t.n] = sgn(a,b);
    t.n++;
  }
  return t;
}
constexpr SpT make_join(){
  SpT t{}; t.n = 0;
  for(int j=0;j<16;j++) for(int k=0;k<16;k++){
    int a = BL[j], b = BL[k];
    int ac = 15^a, bc = 15^b;
    if (ac & bc) continue;
    int mm = ac^bc;
    float s = sgn(a,ac)*sgn(b,bc)*sgn(ac,bc)*sgn(15^mm, mm);
    t.e[t.n] = (unsigned short)(idxof(15^mm) | (j<<4) | (k<<8));
    t.s[t.n] = s;
    t.n++;
  }
  return t;
}
__device__ __constant__ SpT GPT = make_gp();    // 192 nonzeros
__device__ __constant__ SpT JNT = make_join();  //  81 nonzeros

// ---------------------------------------------------------------------------
// helpers
// ---------------------------------------------------------------------------
typedef __bf16 v16bf __attribute__((ext_vector_type(16)));
typedef float  v8f   __attribute__((ext_vector_type(8)));

__device__ inline unsigned short f2bf(float f){
  unsigned int u = __float_as_uint(f);
  unsigned int r = (u + 0x7FFFu + ((u >> 16) & 1u)) >> 16;
  return (unsigned short)r;
}
__device__ inline __bf16 bfbits(unsigned short u){
  union { unsigned short u; __bf16 b; } x; x.u = u; return x.b;
}
__device__ inline float geluf(float x){ return 0.5f*x*(1.0f + erff(x*0.70710678118654752f)); }

// ---------------------------------------------------------------------------
// cross-lane reduction over the 16 lanes sharing a WMMA C-tile row.
// v_permlane16_b32 (VALU) keeps the DS pipe free for tile staging.
// ---------------------------------------------------------------------------
#if __has_builtin(__builtin_amdgcn_permlane16)
#define HAVE_PL16 1
__device__ inline float lanexor16(float v, unsigned sel0, unsigned sel1){
  return __int_as_float(__builtin_amdgcn_permlane16(
      __float_as_int(v), __float_as_int(v), (int)sel0, (int)sel1, false, false));
}
#endif

__device__ inline float rowmax16(float v){
#ifdef HAVE_PL16
  v = fmaxf(v, lanexor16(v, 0x67452301u, 0xEFCDAB89u));  // xor 1
  v = fmaxf(v, lanexor16(v, 0x54761032u, 0xDCFEBA98u));  // xor 2 in lanes 0-7,12-15; lanes 8-11 keep own
  v = fmaxf(v, lanexor16(v, 0x32107654u, 0xBA98FEDCu));  // xor 4
  v = fmaxf(v, lanexor16(v, 0xFEDCBA98u, 0x76543210u));  // xor 8
#else
  for (int msk=1; msk<16; msk<<=1) v = fmaxf(v, __shfl_xor(v, msk, 32));
#endif
  return v;
}
__device__ inline float rowsum16(float v){
#ifdef HAVE_PL16
  v += lanexor16(v, 0x67452301u, 0xEFCDAB89u);
  v += lanexor16(v, 0x54761032u, 0xDCFEBA98u);
  v += lanexor16(v, 0x32107654u, 0xBA98FEDCu);
  v += lanexor16(v, 0xFEDCBA98u, 0x76543210u);
#else
  for (int msk=1; msk<16; msk<<=1) v += __shfl_xor(v, msk, 32);
#endif
  return v;
}

// equi_linear output element: h is LDS [IC][16], W is global (OC,IC,9) row-major
__device__ inline float equi_out(const float* __restrict__ h, int IC,
                                 const float* __restrict__ W, int o, int a){
  int g = GRADE[a];
  const float* w = W + (size_t)o*IC*9;
  float acc = 0.f;
  for (int i=0;i<IC;i++) acc += h[i*16 + a] * w[i*9 + g];
  if (HASE0[a]){
    int pa = PARTNER[a], y = 4 + g;
    for (int i=0;i<IC;i++) acc += h[i*16 + pa] * w[i*9 + y];
  }
  return acc;
}

#define NB 2048
#define HEADS 8
#define QPW 2    // query tiles per wave; 2 waves/block share LDS-staged K/V

// ---------------------------------------------------------------------------
// K1: layernorm + qkv projection; write Q,K [bh][n][64] bf16, V [bh][feat48][n]
// ---------------------------------------------------------------------------
__global__ __launch_bounds__(64) void k_qkv(
    const float* __restrict__ mv, const float* __restrict__ s,
    const float* __restrict__ Wqkv, const float* __restrict__ s2mv,
    const float* __restrict__ s2mv_b, const float* __restrict__ mv2s,
    const float* __restrict__ s2s, const float* __restrict__ s_b,
    unsigned short* __restrict__ Q, unsigned short* __restrict__ K,
    unsigned short* __restrict__ V)
{
  __shared__ float h[256];
  __shared__ float hs[32];
  __shared__ float red[64];
  int t = blockIdx.x, tid = threadIdx.x;
  int b = t >> 11, n = t & (NB-1);
  const float* x = mv + (size_t)t*256;

  float psq = 0.f;
  for (int idx=tid; idx<256; idx+=64){ float v = x[idx]; h[idx] = v; psq += v*v*IPF[idx&15]; }
  red[tid] = psq; __syncthreads();
  for (int off=32; off; off>>=1){ if (tid<off) red[tid]+=red[tid+off]; __syncthreads(); }
  float rms = rsqrtf(red[0]/16.f + 0.01f);
  __syncthreads();

  const float* sv = s + (size_t)t*32;
  float sval = (tid<32) ? sv[tid] : 0.f;
  red[tid] = (tid<32) ? sval : 0.f; __syncthreads();
  for (int off=32; off; off>>=1){ if (tid<off) red[tid]+=red[tid+off]; __syncthreads(); }
  float mu = red[0]/32.f; __syncthreads();
  red[tid] = (tid<32) ? (sval-mu)*(sval-mu) : 0.f; __syncthreads();
  for (int off=32; off; off>>=1){ if (tid<off) red[tid]+=red[tid+off]; __syncthreads(); }
  float var = red[0]/32.f;
  if (tid<32) hs[tid] = (sval-mu)*rsqrtf(var + 1e-5f);
  for (int idx=tid; idx<256; idx+=64) h[idx] *= rms;
  __syncthreads();

  const float qscale = 0.16666667f;  // 1/sqrt(36)
  for (int idx=tid; idx<768; idx+=64){
    int o = idx>>4, a = idx&15;
    float v = equi_out(h, 16, Wqkv, o, a);
    if (a==0){ float acc = s2mv_b[o]; for(int j=0;j<32;j++) acc += hs[j]*s2mv[o*32+j]; v += acc; }
    int which = o>>4, head = (o&15)>>1, cm = o&1;
    int feat = cm*16 + a, bh = b*HEADS + head;
    if      (which==0) Q[((size_t)bh*NB + n)*64 + feat] = f2bf(v * IPF[a] * qscale);
    else if (which==1) K[((size_t)bh*NB + n)*64 + feat] = f2bf(v);
    else               V[((size_t)bh*48 + feat)*NB + n] = f2bf(v);
  }
  for (int o=tid; o<96; o+=64){
    float acc = s_b[o];
    for (int i=0;i<16;i++) acc += h[i*16]*mv2s[o*16+i];
    for (int j=0;j<32;j++) acc += hs[j]*s2s[o*32+j];
    int which = o>>5, head = (o&31)>>2, cs = o&3;
    int feat = 32+cs, bh = b*HEADS + head;
    if      (which==0) Q[((size_t)bh*NB + n)*64 + feat] = f2bf(acc * qscale);
    else if (which==1) K[((size_t)bh*NB + n)*64 + feat] = f2bf(acc);
    else               V[((size_t)bh*48 + feat)*NB + n] = f2bf(acc);
  }
}

// ---------------------------------------------------------------------------
// K2: flash attention. Workgroup = 2 waves sharing one LDS-staged K/V block;
// each wave owns 2 query tiles (32 queries) -> 64-query reuse of global K/V
// with per-wave register footprint < 256 VGPRs (no VGPR-MSB churn).
// ---------------------------------------------------------------------------
__global__ __launch_bounds__(64) void k_attn(
    const unsigned short* __restrict__ Q, const unsigned short* __restrict__ K,
    const unsigned short* __restrict__ V, float* __restrict__ O)
{
  __shared__ unsigned short Kt[32*64];        // [key rel][feat]   4 KB
  __shared__ unsigned short Vt[48*32];        // [feat][key rel]   3 KB
  __shared__ unsigned short Pst[2][16*32];    // per-wave P tile   2 KB
  int blk = blockIdx.x;
  int bh = blk >> 5;                          // 32 query groups (of 64) per bh
  int qg = blk & 31;
  int tid = threadIdx.x;
  int w = tid >> 5, lane = tid & 31;
  int nlo = lane & 15, lh = lane >> 4;
  int q0 = qg*64 + w*32;                      // this wave's first query row
  const unsigned short* Qb = Q + (size_t)bh*NB*64;
  const unsigned short* Kb = K + (size_t)bh*NB*64;
  const unsigned short* Vb = V + (size_t)bh*48*NB;

  // Q as A-matrices (16x32 bf16): A layout K = lh*16+j
  v16bf qa0[QPW], qa1[QPW];
#pragma unroll
  for (int q=0;q<QPW;q++){
    const unsigned short* qr = Qb + (size_t)(q0 + q*16 + nlo)*64;
#pragma unroll
    for (int j=0;j<8;j++){
      qa0[q][j]   = bfbits(qr[lh*16 + j]);
      qa0[q][j+8] = bfbits(qr[lh*16 + 8 + j]);
      qa1[q][j]   = bfbits(qr[32 + lh*16 + j]);
      qa1[q][j+8] = bfbits(qr[32 + lh*16 + 8 + j]);
    }
  }
  v8f acc[QPW][3];
  float mrow[QPW][8], lrow[QPW][8];
#pragma unroll
  for (int q=0;q<QPW;q++){
#pragma unroll
    for (int f=0;f<3;f++) acc[q][f] = (v8f){};
#pragma unroll
    for (int r=0;r<8;r++){ mrow[q][r] = -1e30f; lrow[q][r] = 0.f; }
  }

  for (int kb=0; kb<NB/32; ++kb){
    int key0 = kb*32;
    // ---- cooperative stage: K rows (32 rows, bytes 0-31 and 64-95 of each), V rows (48x64B) ----
    {
      int row = tid >> 1, hf = tid & 1;
      const uint4* ks = (const uint4*)(Kb + (size_t)(key0 + row)*64 + hf*32);
      uint4* kd = (uint4*)(Kt + row*64 + hf*32);
      kd[0] = ks[0]; kd[1] = ks[1];
      if (tid < 48){
        const uint4* vs = (const uint4*)(Vb + (size_t)tid*NB + key0);
        uint4* vd = (uint4*)(Vt + tid*32);
        vd[0]=vs[0]; vd[1]=vs[1]; vd[2]=vs[2]; vd[3]=vs[3];
      }
    }
    __syncthreads();

    // ---- per-wave B-tiles from LDS (contiguous 32B per lane) ----
    v16bf kb00, kb01, kb10, kb11;
    {
      const unsigned short* kr0 = Kt + (size_t)nlo*64 + lh*16;
      const unsigned short* kr1 = Kt + (size_t)(16 + nlo)*64 + lh*16;
#pragma unroll
      for (int j=0;j<16;j++){
        kb00[j] = bfbits(kr0[j]);  kb01[j] = bfbits(kr0[32 + j]);
        kb10[j] = bfbits(kr1[j]);  kb11[j] = bfbits(kr1[32 + j]);
      }
    }
    v16bf vb[3];
#pragma unroll
    for (int f=0; f<3; ++f){
      const unsigned short* vr = Vt + (size_t)(f*16 + nlo)*32 + lh*16;
#pragma unroll
      for (int j=0;j<16;j++) vb[f][j] = bfbits(vr[j]);
    }

#pragma unroll
    for (int q=0; q<QPW; ++q){
      v8f s0 = {}, s1 = {};
      s0 = __builtin_amdgcn_wmma_f32_16x16x32_bf16(false, qa0[q], false, kb00, (short)0, s0, false, false);
      s0 = __builtin_amdgcn_wmma_f32_16x16x32_bf16(false, qa1[q], false, kb01, (short)0, s0, false, false);
      s1 = __builtin_amdgcn_wmma_f32_16x16x32_bf16(false, qa0[q], false, kb10, (short)0, s1, false, false);
      s1 = __builtin_amdgcn_wmma_f32_16x16x32_bf16(false, qa1[q], false, kb11, (short)0, s1, false, false);

      // online softmax (C-tile row = r + 8*lh, col = nlo)
      float p0[8], p1[8];
#pragma unroll
      for (int r=0;r<8;r++){
        float mr = rowmax16(fmaxf(s0[r], s1[r]));
        float mnew = fmaxf(mrow[q][r], mr);
        float al = __expf(mrow[q][r] - mnew);
        p0[r] = __expf(s0[r] - mnew);
        p1[r] = __expf(s1[r] - mnew);
        lrow[q][r] = lrow[q][r]*al + rowsum16(p0[r] + p1[r]);
        mrow[q][r] = mnew;
        acc[q][0][r] *= al; acc[q][1][r] *= al; acc[q][2][r] *= al;
      }
      // stage P (C layout -> LDS row-major 16x32), reload as A-matrix
#pragma unroll
      for (int r=0;r<8;r++){
        int row = r + lh*8;
        Pst[w][row*32 + nlo]      = f2bf(p0[r]);
        Pst[w][row*32 + 16 + nlo] = f2bf(p1[r]);
      }
      __syncthreads();
      v16bf pa;
      {
        const unsigned short* pr = &Pst[w][nlo*32];
#pragma unroll
        for (int j=0;j<8;j++){
          pa[j]   = bfbits(pr[lh*16 + j]);
          pa[j+8] = bfbits(pr[lh*16 + 8 + j]);
        }
      }
      __syncthreads();
      acc[q][0] = __builtin_amdgcn_wmma_f32_16x16x32_bf16(false, pa, false, vb[0], (short)0, acc[q][0], false, false);
      acc[q][1] = __builtin_amdgcn_wmma_f32_16x16x32_bf16(false, pa, false, vb[1], (short)0, acc[q][1], false, false);
      acc[q][2] = __builtin_amdgcn_wmma_f32_16x16x32_bf16(false, pa, false, vb[2], (short)0, acc[q][2], false, false);
    }
    __syncthreads();   // all waves done with Kt/Vt before restaging
  }
  float* Ob = O + (size_t)bh*NB*48;
#pragma unroll
  for (int q=0;q<QPW;q++){
#pragma unroll
    for (int r=0;r<8;r++){
      int row = q*16 + r + lh*8;
      float inv = 1.f / lrow[q][r];
      Ob[(size_t)(q0+row)*48 + nlo]      = acc[q][0][r]*inv;
      Ob[(size_t)(q0+row)*48 + 16 + nlo] = acc[q][1][r]*inv;
      Ob[(size_t)(q0+row)*48 + 32 + nlo] = acc[q][2][r]*inv;
    }
  }
}

// ---------------------------------------------------------------------------
// K3: attention output projection + residual -> x1 into d_out
// ---------------------------------------------------------------------------
__global__ __launch_bounds__(64) void k_attnout(
    const float* __restrict__ mv, const float* __restrict__ s,
    const float* __restrict__ O,
    const float* __restrict__ W, const float* __restrict__ s2mv,
    const float* __restrict__ s2mv_b, const float* __restrict__ mv2s,
    const float* __restrict__ s2s, const float* __restrict__ s_b,
    float* __restrict__ out_mv, float* __restrict__ out_s)
{
  __shared__ float omv[256];
  __shared__ float os[32];
  int t = blockIdx.x, tid = threadIdx.x;
  int b = t >> 11, n = t & (NB-1);
  for (int idx=tid; idx<256; idx+=64){
    int c = idx>>4, a = idx&15, hh = c>>1, cm = c&1;
    omv[idx] = O[((size_t)(b*HEADS+hh)*NB + n)*48 + cm*16 + a];
  }
  if (tid < 32){
    int hh = tid>>2, cs = tid&3;
    os[tid] = O[((size_t)(b*HEADS+hh)*NB + n)*48 + 32 + cs];
  }
  __syncthreads();
  for (int idx=tid; idx<256; idx+=64){
    int o = idx>>4, a = idx&15;
    float v = equi_out(omv, 16, W, o, a);
    if (a==0){ float acc = s2mv_b[o]; for(int j=0;j<32;j++) acc += os[j]*s2mv[o*32+j]; v += acc; }
    out_mv[(size_t)t*256 + idx] = mv[(size_t)t*256 + idx] + v;
  }
  if (tid < 32){
    int o = tid;
    float acc = s_b[o];
    for (int i=0;i<16;i++) acc += omv[i*16]*mv2s[o*16+i];
    for (int j=0;j<32;j++) acc += os[j]*s2s[o*32+j];
    out_s[(size_t)t*32 + o] = s[(size_t)t*32 + o] + acc;
  }
}

// ---------------------------------------------------------------------------
// K4: geometric MLP (norm, bilinear GP/JOIN, gated GELU, out) + residual
// ---------------------------------------------------------------------------
__global__ __launch_bounds__(64) void k_mlp(
    const float* __restrict__ ref,
    const float* __restrict__ blw, const float* __restrict__ bls2mv, const float* __restrict__ blb,
    const float* __restrict__ brw, const float* __restrict__ brs2mv, const float* __restrict__ brb,
    const float* __restrict__ bow, const float* __restrict__ bos2mv, const float* __restrict__ bos2mv_b,
    const float* __restrict__ bomv2s, const float* __restrict__ bos2s, const float* __restrict__ bosb,
    const float* __restrict__ mow, const float* __restrict__ mos2mv, const float* __restrict__ mos2mv_b,
    const float* __restrict__ momv2s, const float* __restrict__ mos2s, const float* __restrict__ mosb,
    float* __restrict__ out_mv, float* __restrict__ out_s)
{
  __shared__ float h[256];
  __shared__ float hs[32];
  __shared__ float red[64];
  __shared__ float lft[512], rgt[512], hid[512], bmv[512];
  __shared__ float bs[64], gate[32], gs[64];
  int t = blockIdx.x, tid = threadIdx.x;
  int b = t >> 11;
  const float* x  = out_mv + (size_t)t*256;   // x1 from K3
  const float* xs = out_s  + (size_t)t*32;

  float psq = 0.f;
  for (int idx=tid; idx<256; idx+=64){ float v = x[idx]; h[idx] = v; psq += v*v*IPF[idx&15]; }
  red[tid] = psq; __syncthreads();
  for (int off=32; off; off>>=1){ if (tid<off) red[tid]+=red[tid+off]; __syncthreads(); }
  float rms = rsqrtf(red[0]/16.f + 0.01f);
  __syncthreads();
  float sval = (tid<32) ? xs[tid] : 0.f;
  red[tid] = (tid<32) ? sval : 0.f; __syncthreads();
  for (int off=32; off; off>>=1){ if (tid<off) red[tid]+=red[tid+off]; __syncthreads(); }
  float mu = red[0]/32.f; __syncthreads();
  red[tid] = (tid<32) ? (sval-mu)*(sval-mu) : 0.f; __syncthreads();
  for (int off=32; off; off>>=1){ if (tid<off) red[tid]+=red[tid+off]; __syncthreads(); }
  float var = red[0]/32.f;
  if (tid<32) hs[tid] = (sval-mu)*rsqrtf(var + 1e-5f);
  for (int idx=tid; idx<256; idx+=64) h[idx] *= rms;
  __syncthreads();

  for (int idx=tid; idx<512; idx+=64){
    int o = idx>>4, a = idx&15;
    float lv = equi_out(h, 16, blw, o, a);
    float rv = equi_out(h, 16, brw, o, a);
    if (a==0){
      float la = blb[o], ra = brb[o];
      for (int j=0;j<32;j++){ la += hs[j]*bls2mv[o*32+j]; ra += hs[j]*brs2mv[o*32+j]; }
      lv += la; rv += ra;
    }
    lft[idx] = lv; rgt[idx] = rv;
  }
  __syncthreads();

  float refw = ref[b*16 + 14];                // e123 component
  if (tid < 32){
    int c = tid & 15;
    float acc[16]; for (int i=0;i<16;i++) acc[i] = 0.f;
    if (tid < 16){
      const float* L = lft + c*16; const float* R = rgt + c*16;
      for (int e=0;e<GPT.n;e++){
        int pk = GPT.e[e];
        acc[pk & 15] += GPT.s[e] * L[(pk>>4)&15] * R[(pk>>8)&15];
      }
      for (int i=0;i<16;i++) hid[c*16+i] = acc[i];
    } else {
      const float* L = lft + (16+c)*16; const float* R = rgt + (16+c)*16;
      for (int e=0;e<JNT.n;e++){
        int pk = JNT.e[e];
        acc[pk & 15] += JNT.s[e] * L[(pk>>4)&15] * R[(pk>>8)&15];
      }
      for (int i=0;i<16;i++) hid[(16+c)*16+i] = refw*acc[i];
    }
  }
  __syncthreads();

  for (int idx=tid; idx<512; idx+=64){
    int o = idx>>4, a = idx&15;
    float v = equi_out(hid, 32, bow, o, a);
    if (a==0){ float acc = bos2mv_b[o]; for(int j=0;j<32;j++) acc += hs[j]*bos2mv[o*32+j]; v += acc; }
    bmv[idx] = v;
  }
  {
    int o = tid;
    float acc = bosb[o];
    for (int i=0;i<32;i++) acc += hid[i*16]*bomv2s[o*32+i];
    for (int j=0;j<32;j++) acc += hs[j]*bos2s[o*32+j];
    bs[o] = acc;
  }
  __syncthreads();
  if (tid < 32) gate[tid] = geluf(bmv[tid*16]);
  gs[tid] = geluf(bs[tid]);
  __syncthreads();
  for (int idx=tid; idx<512; idx+=64) lft[idx] = gate[idx>>4]*bmv[idx];  // g_mv
  __syncthreads();

  for (int idx=tid; idx<256; idx+=64){
    int o = idx>>4, a = idx&15;
    float v = equi_out(lft, 32, mow, o, a);
    if (a==0){ float acc = mos2mv_b[o]; for(int j=0;j<64;j++) acc += gs[j]*mos2mv[o*64+j]; v += acc; }
    out_mv[(size_t)t*256 + idx] = x[idx] + v;
  }
  if (tid < 32){
    int o = tid;
    float acc = mosb[o];
    for (int i=0;i<32;i++) acc += lft[i*16]*momv2s[o*32+i];
    for (int j=0;j<64;j++) acc += gs[j]*mos2s[o*64+j];
    out_s[(size_t)t*32 + o] = xs[o] + acc;
  }
}

// ---------------------------------------------------------------------------
// launch
// ---------------------------------------------------------------------------
extern "C" void kernel_launch(void* const* d_in, const int* in_sizes, int n_in,
                              void* d_out, int out_size, void* d_ws, size_t ws_size,
                              hipStream_t stream) {
  const float* mv  = (const float*)d_in[0];
  const float* s   = (const float*)d_in[1];
  const float* ref = (const float*)d_in[2];
  const float* qkv_w      = (const float*)d_in[3];
  const float* qkv_s2mv   = (const float*)d_in[4];
  const float* qkv_s2mv_b = (const float*)d_in[5];
  const float* qkv_mv2s   = (const float*)d_in[6];
  const float* qkv_s2s    = (const float*)d_in[7];
  const float* qkv_s_b    = (const float*)d_in[8];
  const float* ao_w       = (const float*)d_in[9];
  const float* ao_s2mv    = (const float*)d_in[10];
  const float* ao_s2mv_b  = (const float*)d_in[11];
  const float* ao_mv2s    = (const float*)d_in[12];
  const float* ao_s2s     = (const float*)d_in[13];
  const float* ao_s_b     = (const float*)d_in[14];
  const float* bl_w       = (const float*)d_in[15];
  const float* bl_s2mv    = (const float*)d_in[16];
  const float* bl_b       = (const float*)d_in[17];
  const float* br_w       = (const float*)d_in[18];
  const float* br_s2mv    = (const float*)d_in[19];
  const float* br_b       = (const float*)d_in[20];
  const float* bo_w       = (const float*)d_in[21];
  const float* bo_s2mv    = (const float*)d_in[22];
  const float* bo_s2mv_b  = (const float*)d_in[23];
  const float* bo_mv2s    = (const float*)d_in[24];
  const float* bo_s2s     = (const float*)d_in[25];
  const float* bo_s_b     = (const float*)d_in[26];
  const float* mo_w       = (const float*)d_in[27];
  const float* mo_s2mv    = (const float*)d_in[28];
  const float* mo_s2mv_b  = (const float*)d_in[29];
  const float* mo_mv2s    = (const float*)d_in[30];
  const float* mo_s2s     = (const float*)d_in[31];
  const float* mo_s_b     = (const float*)d_in[32];

  float* out_mv = (float*)d_out;                       // (2,2048,16,16)
  float* out_s  = out_mv + (size_t)2*NB*16*16;         // (2,2048,32)

  const size_t BH = 2*HEADS;
  unsigned short* Q = (unsigned short*)d_ws;           // [bh][n][64] bf16
  unsigned short* K = Q + BH*NB*64;                    // [bh][n][64] bf16
  unsigned short* V = K + BH*NB*64;                    // [bh][feat48][n] bf16
  float*          O = (float*)(V + BH*NB*48);          // [bh][n][48] f32

  // zero Q/K/V so the d-padding (feats 36..63 / 36..47) is exact zero
  hipMemsetAsync(d_ws, 0, (size_t)(BH*NB*64*2*2 + BH*NB*48*2), stream);

  k_qkv<<<2*NB, 64, 0, stream>>>(mv, s, qkv_w, qkv_s2mv, qkv_s2mv_b,
                                 qkv_mv2s, qkv_s2s, qkv_s_b, Q, K, V);
  k_attn<<<(int)(BH*(NB/64)), 64, 0, stream>>>(Q, K, V, O);
  k_attnout<<<2*NB, 64, 0, stream>>>(mv, s, O, ao_w, ao_s2mv, ao_s2mv_b,
                                     ao_mv2s, ao_s2s, ao_s_b, out_mv, out_s);
  k_mlp<<<2*NB, 64, 0, stream>>>(ref,
      bl_w, bl_s2mv, bl_b, br_w, br_s2mv, br_b,
      bo_w, bo_s2mv, bo_s2mv_b, bo_mv2s, bo_s2s, bo_s_b,
      mo_w, mo_s2mv, mo_s2mv_b, mo_mv2s, mo_s2s, mo_s_b,
      out_mv, out_s);
}
